// Net_MNIST_66760971649173
// MI455X (gfx1250) — matrix operand order repaired and checked by reading, not compiled, not run on hardware
//
#include <hip/hip_runtime.h>
#include <cstddef>

#define BATCH  32768
#define D_IN   784
#define KP     832      // D padded to multiple of 64 (13 K-tiles of 64)
#define H      500
#define HP     512      // H padded to multiple of 64
#define NLAB   10
#define EPS_   1e-4f

typedef __attribute__((ext_vector_type(16))) __bf16 v16bf;
typedef __attribute__((ext_vector_type(8)))  float  v8f;

union Frag16 { v16bf bf; unsigned int w[8]; };

__device__ __forceinline__ unsigned short f2bf(float f) {
  unsigned int u = __float_as_uint(f);
  unsigned int r = u + 0x7FFFu + ((u >> 16) & 1u);   // round-to-nearest-even
  return (unsigned short)(r >> 16);
}
__device__ __forceinline__ float bf2f(unsigned short h) {
  return __uint_as_float(((unsigned int)h) << 16);
}

__device__ __forceinline__ v8f wmma_bf16(v16bf a, v16bf b, v8f c) {
  return __builtin_amdgcn_wmma_f32_16x16x32_bf16(false, a, false, b, (short)0, c, false, false);
}

// B fragment: weights [N][K] row-major bf16. Lane: N = n0 + (lane&15),
// elems 0..7 = K k0+kh*8.., elems 8..15 = K k0+16+kh*8.. (kh = lane>>4) -> two 16B loads.
__device__ __forceinline__ v16bf load_b_frag(const unsigned short* base, int rowstride,
                                             int n0, int k0) {
  int lane = threadIdx.x & 31;
  const unsigned short* p = base + (size_t)(n0 + (lane & 15)) * rowstride
                                 + k0 + ((lane >> 4) << 3);
  uint4 lo = ((const uint4*)p)[0], hi = ((const uint4*)(p + 16))[0];
  Frag16 f;
  f.w[0] = lo.x; f.w[1] = lo.y; f.w[2] = lo.z; f.w[3] = lo.w;
  f.w[4] = hi.x; f.w[5] = hi.y; f.w[6] = hi.z; f.w[7] = hi.w;
  return f.bf;
}

// A fragment from LDS tile [M][rowstride] bf16. Lane: M = m0 + (lane&15),
// elems 0..7 = K kh*8.., elems 8..15 = K 16+kh*8.. (kh = lane>>4).
__device__ __forceinline__ v16bf load_a_frag_lds(const unsigned short* L, int rowstride,
                                                 int m0, int k0) {
  int lane = threadIdx.x & 31;
  const unsigned short* row = L + (size_t)(m0 + (lane & 15)) * rowstride + k0;
  int kh = (lane >> 4) * 8;
  const uint4* q0 = (const uint4*)(row + kh);
  const uint4* q1 = (const uint4*)(row + 16 + kh);
  uint4 lo = q0[0], hi = q1[0];
  Frag16 f;
  f.w[0] = lo.x; f.w[1] = lo.y; f.w[2] = lo.z; f.w[3] = lo.w;
  f.w[4] = hi.x; f.w[5] = hi.y; f.w[6] = hi.z; f.w[7] = hi.w;
  return f.bf;
}

// ---------------- prep kernels ----------------

__global__ void k_init(unsigned* xb) {
  if (threadIdx.x == 0 && blockIdx.x == 0) *xb = 0u;
}

__global__ void k_max(const float* __restrict__ x, unsigned* xb, int n) {
  float m = 0.f;
  for (int i = blockIdx.x * blockDim.x + threadIdx.x; i < n; i += gridDim.x * blockDim.x)
    m = fmaxf(m, x[i]);
  for (int off = 16; off; off >>= 1) m = fmaxf(m, __shfl_xor(m, off));
  if ((threadIdx.x & 31) == 0) atomicMax(xb, __float_as_uint(m)); // x >= 0
}

// x (fp32) -> bf16, label head (cols<10) and pad (>=784) zeroed. One row per block.
__global__ __launch_bounds__(256) void k_cvt_x(const float* __restrict__ x,
                                               unsigned short* __restrict__ xbf) {
  int r = blockIdx.x;
  const float* xr = x + (size_t)r * D_IN;
  unsigned short* o = xbf + (size_t)r * KP;
  for (int k = threadIdx.x; k < KP; k += 256) {
    float v = (k >= NLAB && k < D_IN) ? xr[k] : 0.f;
    o[k] = f2bf(v);
  }
}

__global__ void k_cvt_w1(const float* __restrict__ W1, unsigned short* __restrict__ W1bf) {
  int i = blockIdx.x * 256 + threadIdx.x;   // 512*832 elems
  int h = i / KP, k = i % KP;
  unsigned short v = 0;
  if (h < H && k < D_IN) v = f2bf(W1[(size_t)h * D_IN + k]);
  W1bf[i] = v;
}

__global__ void k_cvt_w2(const float* __restrict__ W2, unsigned short* __restrict__ W2bf) {
  int i = blockIdx.x * 256 + threadIdx.x;   // 512*512 elems
  int n = i / HP, k = i % HP;
  unsigned short v = 0;
  if (n < H && k < H) v = f2bf(W2[(size_t)n * H + k]);
  W2bf[i] = v;
}

// W1ht[lab][h] = xmax * W1[h][lab]  (contiguous per-label stream for k_label)
__global__ void k_cvt_w1h(const float* __restrict__ W1, const unsigned* xb,
                          float* __restrict__ W1ht) {
  int lab = blockIdx.x;     // 10
  int h = threadIdx.x;      // 512
  float xmax = __uint_as_float(*xb);
  W1ht[lab * HP + h] = (h < H) ? xmax * W1[(size_t)h * D_IN + lab] : 0.f;
}

// wave per row, lane-strided, shuffle reduce
__global__ __launch_bounds__(256) void k_rownorm(const float* __restrict__ x,
                                                 const unsigned* xb,
                                                 float* __restrict__ inv_n1) {
  int wave = threadIdx.x >> 5, lane = threadIdx.x & 31;
  int r = blockIdx.x * 8 + wave;
  const float* xr = x + (size_t)r * D_IN;
  float s = 0.f;
  for (int i = NLAB + lane; i < D_IN; i += 32) s = fmaf(xr[i], xr[i], s);
  for (int off = 16; off; off >>= 1) s += __shfl_xor(s, off);
  if (lane == 0) {
    float xmax = __uint_as_float(*xb);
    inv_n1[r] = 1.f / (sqrtf(fmaf(xmax, xmax, s)) + EPS_);
  }
}

// ---------------- GEMM1: T = x_pad @ W1_pad^T (label-independent) ----------------
// block 256 thr (8 waves), tile M=128 x N=64, K-tile 64 (2 wmma-K per tile),
// double-buffered LDS A staged with async global->LDS copies.
#define AST 72   // padded LDS row stride (bf16 elems): 144B, conflict-free

__global__ __launch_bounds__(256) void k_gemm1(const unsigned short* __restrict__ xbf,
                                               const unsigned short* __restrict__ W1bf,
                                               unsigned short* __restrict__ T) {
  __shared__ unsigned short As[2][128 * AST];
  const int tid  = threadIdx.x;
  const int lane = tid & 31;
  const int wave = tid >> 5;
  const int b0 = blockIdx.x * 128;
  const int n0 = blockIdx.y * 64;
  const int wm = wave & 3;
  const int wn = wave >> 2;
  v8f acc[2][2] = {};

  // async stage one 128x64 bf16 tile: 1024 chunks of 16B, 4 per thread
  auto stage = [&](int buf, int k0) {
#pragma unroll
    for (int i = 0; i < 4; ++i) {
      int q  = tid + i * 256;
      int rr = q >> 3;              // row 0..127
      int c8 = (q & 7);             // 16B chunk within row
      const unsigned short* g = xbf + (size_t)(b0 + rr) * KP + k0 + c8 * 8;
      unsigned l = (unsigned)(size_t)(&As[buf][rr * AST + c8 * 8]);
      asm volatile("global_load_async_to_lds_b128 %0, %1, off"
                   :: "v"(l), "v"(g) : "memory");
    }
  };

  stage(0, 0);
  asm volatile("s_wait_asynccnt 0x0" ::: "memory");
  __syncthreads();

  for (int k0 = 0; k0 < KP; k0 += 64) {
    const int buf = (k0 >> 6) & 1;
    if (k0 + 64 < KP) stage(buf ^ 1, k0 + 64);   // prefetch next tile
#pragma unroll
    for (int kk = 0; kk < 64; kk += 32) {
      v16bf a0 = load_a_frag_lds(&As[buf][0], AST, wm * 32,      kk);
      v16bf a1 = load_a_frag_lds(&As[buf][0], AST, wm * 32 + 16, kk);
      v16bf bA = load_b_frag(W1bf, KP, n0 + wn * 32,      k0 + kk);
      v16bf bB = load_b_frag(W1bf, KP, n0 + wn * 32 + 16, k0 + kk);
      acc[0][0] = wmma_bf16(a0, bA, acc[0][0]);
      acc[0][1] = wmma_bf16(a0, bB, acc[0][1]);
      acc[1][0] = wmma_bf16(a1, bA, acc[1][0]);
      acc[1][1] = wmma_bf16(a1, bB, acc[1][1]);
    }
    asm volatile("s_wait_asynccnt 0x0" ::: "memory");
    __syncthreads();
  }

#pragma unroll
  for (int am = 0; am < 2; ++am)
#pragma unroll
    for (int bn = 0; bn < 2; ++bn) {
      int Mb = b0 + wm * 32 + am * 16 + (lane >> 4) * 8;
      int Nc = n0 + wn * 32 + bn * 16 + (lane & 15);
#pragma unroll
      for (int rr = 0; rr < 8; ++rr)
        T[(size_t)(Mb + rr) * HP + Nc] = f2bf(acc[am][bn][rr]);
    }
}

// ---------------- per-label: layer1 epilogue + layer2 GEMM + goodness ----------------
// grid (B/64, 10); block 256 thr (8 waves). Each block: 64 rows, one label.
__global__ __launch_bounds__(256) void k_label(const unsigned short* __restrict__ T,
                                               const unsigned short* __restrict__ W2bf,
                                               const float* __restrict__ W1ht,
                                               const float* __restrict__ b1,
                                               const float* __restrict__ b2,
                                               const float* __restrict__ inv_n1,
                                               float* __restrict__ out) {
  __shared__ unsigned short Z[64 * 520];
  __shared__ float g1buf[64], g2buf[64], invb[64];
  const int tid  = threadIdx.x;
  const int lane = tid & 31;
  const int wave = tid >> 5;
  const int b0  = blockIdx.x * 64;
  const int lab = blockIdx.y;
  const float* W1h = W1ht + lab * HP;

  if (tid < 64) { g1buf[tid] = 0.f; g2buf[tid] = 0.f; }
  __syncthreads();

  // layer-1 epilogue: z1 = relu((T + xmax*W1[:,lab]) * inv_n1 + b1)
  const int rr0 = tid >> 2;
  const int qq  = tid & 3;
  {
    const unsigned short* Trow = T + (size_t)(b0 + rr0) * HP;
    const float inv1 = inv_n1[b0 + rr0];
    const int cbeg = qq * 128, cend = cbeg + 128;
    const int cval = cend < H ? cend : H;     // split instead of per-elem guard
    float s = 0.f;
    for (int c = cbeg; c < cval; ++c) {
      float pre = fmaf(bf2f(Trow[c]) + W1h[c], inv1, b1[c]);
      float z = fmaxf(pre, 0.f);
      s = fmaf(z, z, s);
      Z[rr0 * 520 + c] = f2bf(z);
    }
    for (int c = cval; c < cend; ++c) Z[rr0 * 520 + c] = 0;
    atomicAdd(&g1buf[rr0], s);
  }
  __syncthreads();
  if (tid < 64) invb[tid] = 1.f / (sqrtf(g1buf[tid]) + EPS_);
  __syncthreads();
  {
    const float iv = invb[rr0];
    for (int c = qq * 128; c < qq * 128 + 128; ++c)
      Z[rr0 * 520 + c] = f2bf(bf2f(Z[rr0 * 520 + c]) * iv);
  }
  __syncthreads();

  // layer-2 GEMM: 64 x 512 per block; each wave owns 64 output cols
  const int n0 = wave * 64;
  v8f acc[4][4] = {};
  for (int k0 = 0; k0 < HP; k0 += 32) {
    v16bf a[4], b[4];
#pragma unroll
    for (int mt = 0; mt < 4; ++mt) a[mt] = load_a_frag_lds(Z, 520, mt * 16, k0);
#pragma unroll
    for (int nt = 0; nt < 4; ++nt) b[nt] = load_b_frag(W2bf, HP, n0 + nt * 16, k0);
    if (k0 + 32 < HP)
      __builtin_prefetch(W2bf + (size_t)(n0 + (lane & 15)) * HP + k0 + 32, 0, 0);
#pragma unroll
    for (int mt = 0; mt < 4; ++mt)
#pragma unroll
      for (int nt = 0; nt < 4; ++nt)
        acc[mt][nt] = wmma_bf16(a[mt], b[nt], acc[mt][nt]);
  }

  // z2 = relu(acc + b2); accumulate sum z2^2 per row (z2 never stored)
#pragma unroll
  for (int mt = 0; mt < 4; ++mt) {
    float rowsum[8] = {};
#pragma unroll
    for (int nt = 0; nt < 4; ++nt) {
      int n = n0 + nt * 16 + (lane & 15);
      bool ok = n < H;
      float bb = ok ? b2[n] : 0.f;
#pragma unroll
      for (int rr = 0; rr < 8; ++rr) {
        float z = ok ? fmaxf(acc[mt][nt][rr] + bb, 0.f) : 0.f;
        rowsum[rr] = fmaf(z, z, rowsum[rr]);
      }
    }
    int mbase = mt * 16 + (lane >> 4) * 8;
#pragma unroll
    for (int rr = 0; rr < 8; ++rr) atomicAdd(&g2buf[mbase + rr], rowsum[rr]);
  }
  __syncthreads();

  if (tid < 64)
    out[(size_t)(b0 + tid) * NLAB + lab] = (g1buf[tid] + g2buf[tid]) * (1.0f / (float)H);
}

// ---------------- launch ----------------

extern "C" void kernel_launch(void* const* d_in, const int* in_sizes, int n_in,
                              void* d_out, int out_size, void* d_ws, size_t ws_size,
                              hipStream_t stream) {
  const float* x  = (const float*)d_in[0];
  const float* W1 = (const float*)d_in[1];
  const float* b1 = (const float*)d_in[2];
  const float* W2 = (const float*)d_in[3];
  const float* b2 = (const float*)d_in[4];
  float* out = (float*)d_out;

  char* ws = (char*)d_ws;
  unsigned*       xb     = (unsigned*)(ws + 0);
  float*          inv_n1 = (float*)(ws + 256);        // 32768*4
  unsigned short* W1bf   = (unsigned short*)(ws + 131328);    // 512*832 bf16
  unsigned short* W2bf   = (unsigned short*)(ws + 983296);    // 512*512 bf16
  float*          W1ht   = (float*)(ws + 1507584);            // 10*512 f32
  unsigned short* T      = (unsigned short*)(ws + 1528064);   // 32768*512 bf16
  unsigned short* xbf    = (unsigned short*)(ws + 35082496);  // 32768*832 bf16

  k_init<<<1, 1, 0, stream>>>(xb);
  k_max<<<2048, 256, 0, stream>>>(x, xb, BATCH * D_IN);
  k_cvt_x<<<BATCH, 256, 0, stream>>>(x, xbf);
  k_cvt_w1<<<(HP * KP) / 256, 256, 0, stream>>>(W1, W1bf);
  k_cvt_w2<<<(HP * HP) / 256, 256, 0, stream>>>(W2, W2bf);
  k_cvt_w1h<<<NLAB, HP, 0, stream>>>(W1, xb, W1ht);
  k_rownorm<<<BATCH / 8, 256, 0, stream>>>(x, xb, inv_n1);
  k_gemm1<<<dim3(BATCH / 128, HP / 64), 256, 0, stream>>>(xbf, W1bf, T);
  k_label<<<dim3(BATCH / 64, NLAB), 256, 0, stream>>>(T, W2bf, W1ht, b1, b2, inv_n1, out);
}
